// LSTMEncoder_64123861729641
// MI455X (gfx1250) — matrix operand order repaired and checked by reading, not compiled, not run on hardware
//
#include <hip/hip_runtime.h>
#include <hip/hip_bf16.h>
#include <math.h>

// B=32, S=1024, E=512, H=1024, DK=DV=512
typedef __bf16 bhalf;
typedef bhalf bhalf16 __attribute__((ext_vector_type(16)));
typedef float floatx8 __attribute__((ext_vector_type(8)));

union FragB16 { int4 q[2]; bhalf16 v; };
union Pack8   { int4 q; unsigned short s[8]; };

__device__ __forceinline__ unsigned short f2bf(float f) {
  unsigned u = __float_as_uint(f);
  u += 0x7FFFu + ((u >> 16) & 1u);       // round-to-nearest-even
  return (unsigned short)(u >> 16);
}
__device__ __forceinline__ float bf2f(unsigned short h) {
  return __uint_as_float(((unsigned)h) << 16);
}
__device__ __forceinline__ float sigmoidf_(float x) {
  return 1.0f / (1.0f + __expf(-x));
}

// ---------------------------------------------------------------- convert
__global__ void k_f32_to_bf16(const float* __restrict__ src,
                              unsigned short* __restrict__ dst, int n) {
  int i = blockIdx.x * blockDim.x + threadIdx.x;
  int stride = gridDim.x * blockDim.x;
  for (; i < n; i += stride) dst[i] = f2bf(src[i]);
}

__global__ void k_init(unsigned* p) { if (threadIdx.x < 2) p[threadIdx.x] = 0u; }

// ---------------------------------------------------------------- phase 1:
// xg[s][n][b] = sum_k x[b,s,k] * W_ih[n,k] + b_ih[n] + b_hh[n]   (bf16 out)
__global__ __launch_bounds__(256) void k_gemm_xg(
    const unsigned short* __restrict__ X,    // [32768][512] bf16
    const unsigned short* __restrict__ W,    // [4096][512]  bf16
    const float* __restrict__ b_ih, const float* __restrict__ b_hh,
    unsigned short* __restrict__ XG)         // [1024][4096][32] bf16
{
  const int lane = threadIdx.x & 31;
  const int lm = lane & 15, lh = lane >> 4;
  const int gw = blockIdx.x * 8 + (threadIdx.x >> 5);
  const int mt = gw >> 6;            // 0..2047
  const int ng = gw & 63;            // 0..63
  const int r0 = mt << 4;
  const int nb = ng << 6;

  floatx8 acc[4] = {};
  const unsigned short* ap0 = X + (size_t)(r0 + lm) * 512;
  for (int k0 = 0; k0 < 512; k0 += 32) {
    FragB16 a;
    const unsigned short* ap = ap0 + k0 + lh * 8;
    a.q[0] = *(const int4*)ap;
    a.q[1] = *(const int4*)(ap + 16);
#pragma unroll
    for (int tt = 0; tt < 4; ++tt) {
      FragB16 b;
      const unsigned short* bp = W + (size_t)(nb + tt * 16 + lm) * 512 + k0 + lh * 8;
      b.q[0] = *(const int4*)bp;
      b.q[1] = *(const int4*)(bp + 16);
      acc[tt] = __builtin_amdgcn_wmma_f32_16x16x32_bf16(
          false, a.v, false, b.v, (short)0, acc[tt], false, false);
    }
  }
#pragma unroll
  for (int tt = 0; tt < 4; ++tt) {
    int n = nb + tt * 16 + lm;
    float bias = b_ih[n] + b_hh[n];
#pragma unroll
    for (int v = 0; v < 8; ++v) {
      int r = r0 + lh * 8 + v;          // flattened (b*1024 + s)
      int bb = r >> 10, s = r & 1023;
      XG[((size_t)s * 4096 + n) * 32 + bb] = f2bf(acc[tt][v] + bias);
    }
  }
}

// ---------------------------------------------------------------- phase 2:
// persistent LSTM scan: 32 blocks x 128 threads = 128 waves.
// Block owns 32 h-cols; its W_hh slice (4g x 32n x 1024k bf16 = 256 KB) is
// async-DMA'd into LDS once (XOR-swizzled 16B chunks to kill bank conflicts)
// and reused for all 1024 timesteps. h(t-1) fragments stream from L2.
__global__ __launch_bounds__(128) void k_lstm_scan(
    const unsigned short* __restrict__ XG,   // [1024][4096][32]
    const unsigned short* __restrict__ Whh,  // [4096][1024] bf16
    unsigned short* __restrict__ HS,         // [32][1024][1024] bf16 (b,t,h)
    unsigned* __restrict__ bar)              // bar[0]=cnt, bar[1]=gen
{
  __shared__ unsigned short ldsW[2 * 4 * 16 * 1024];  // 256 KB
  const int tid  = threadIdx.x;
  const int lane = tid & 31;
  const int lm = lane & 15, lh = lane >> 4;
  const int w   = tid >> 5;            // wave in block: 0..3
  const int bg  = w & 1;               // batch group
  const int hgl = w >> 1;              // local h-col group: 0..1
  const int b0  = bg << 4;
  const int n0  = (blockIdx.x * 2 + hgl) << 4;

  // ---- async preload of W_hh slice into LDS (16384 x 16B chunks)
  for (int L = tid; L < 16384; L += 128) {
    int rowId = L >> 7;                 // (hgl2*4+g)*16+row, 0..127
    int cc    = L & 127;                // 16B chunk within 2KB row
    int hg2   = rowId >> 6;
    int g     = (rowId >> 4) & 3;
    int row   = rowId & 15;
    int grow  = g * 1024 + (blockIdx.x * 2 + hg2) * 16 + row;
    unsigned long long gaddr =
        (unsigned long long)(uintptr_t)(Whh + (size_t)grow * 1024 + cc * 8);
    unsigned laddr =
        (unsigned)(uintptr_t)&ldsW[rowId * 1024 + ((cc ^ row) << 3)];
    asm volatile("global_load_async_to_lds_b128 %0, %1, off"
                 :: "v"(laddr), "v"(gaddr) : "memory");
  }
  asm volatile("s_wait_asynccnt 0x0" ::: "memory");
  __syncthreads();

  const unsigned short* wbase0 = ldsW + (size_t)(hgl * 4) * 16 * 1024;

  floatx8 c = {};                       // cell state, lives in registers

  for (int t = 0; t < 1024; ++t) {
    floatx8 g4[4] = {};
    if (t > 0) {
      const unsigned short* arow =
          HS + ((size_t)(b0 + lm) * 1024 + (t - 1)) * 1024;
      for (int k0 = 0; k0 < 1024; k0 += 32) {
        FragB16 a;
        a.q[0] = *(const int4*)(arow + k0 + lh * 8);
        a.q[1] = *(const int4*)(arow + k0 + lh * 8 + 16);
        int c0 = (k0 >> 3) + lh;        // 16B chunk index within row
#pragma unroll
        for (int g = 0; g < 4; ++g) {
          const unsigned short* wrow = wbase0 + (size_t)(g * 16 + lm) * 1024;
          FragB16 bfr;
          bfr.q[0] = *(const int4*)(wrow + ((c0 ^ lm) << 3));
          bfr.q[1] = *(const int4*)(wrow + (((c0 + 2) ^ lm) << 3));
          g4[g] = __builtin_amdgcn_wmma_f32_16x16x32_bf16(
              false, a.v, false, bfr.v, (short)0, g4[g], false, false);
        }
      }
    }

    // ---- add precomputed input contribution xg(t) (includes biases)
#pragma unroll
    for (int g = 0; g < 4; ++g) {
      Pack8 p;
      p.q = *(const int4*)(XG + ((size_t)t * 4096 + g * 1024 + n0 + lm) * 32 +
                           b0 + lh * 8);
#pragma unroll
      for (int v = 0; v < 8; ++v) g4[g][v] += bf2f(p.s[v]);
    }

    // ---- elementwise LSTM cell + store h(t)
#pragma unroll
    for (int v = 0; v < 8; ++v) {
      float iv = sigmoidf_(g4[0][v]);
      float fv = sigmoidf_(g4[1][v]);
      float gv = tanhf(g4[2][v]);
      float ov = sigmoidf_(g4[3][v]);
      float cv = fv * c[v] + iv * gv;
      c[v] = cv;
      float hv = ov * tanhf(cv);
      int b = b0 + lh * 8 + v;
      HS[((size_t)b * 1024 + t) * 1024 + n0 + lm] = f2bf(hv);
    }

    // ---- device-wide barrier before next step (32 blocks)
    __threadfence();
    __syncthreads();
    if (tid == 0) {
      unsigned gen = __hip_atomic_load(&bar[1], __ATOMIC_ACQUIRE,
                                       __HIP_MEMORY_SCOPE_AGENT);
      unsigned arr = __hip_atomic_fetch_add(&bar[0], 1u, __ATOMIC_ACQ_REL,
                                            __HIP_MEMORY_SCOPE_AGENT);
      if (arr == 31u) {
        __hip_atomic_store(&bar[0], 0u, __ATOMIC_RELAXED,
                           __HIP_MEMORY_SCOPE_AGENT);
        __hip_atomic_store(&bar[1], gen + 1u, __ATOMIC_RELEASE,
                           __HIP_MEMORY_SCOPE_AGENT);
      } else {
        while (__hip_atomic_load(&bar[1], __ATOMIC_ACQUIRE,
                                 __HIP_MEMORY_SCOPE_AGENT) == gen) {
          __builtin_amdgcn_s_sleep(2);
        }
      }
    }
    __syncthreads();
  }
}

// ---------------------------------------------------------------- phase 3:
// out[r][n] = sum_k hs[r][k] * W[n][k] + bias[n];  M=32768, N=512, K=1024
__global__ __launch_bounds__(256) void k_gemm_out(
    const unsigned short* __restrict__ HS,  // [32768][1024] bf16
    const unsigned short* __restrict__ W,   // [512][1024]  bf16
    const float* __restrict__ bias,         // [512]
    float* __restrict__ out)                // [32768][512] f32
{
  const int lane = threadIdx.x & 31;
  const int lm = lane & 15, lh = lane >> 4;
  const int gw = blockIdx.x * 8 + (threadIdx.x >> 5);
  const int mt = gw >> 3;        // 0..2047
  const int ng = gw & 7;         // 0..7
  const int r0 = mt << 4;
  const int nb = ng << 6;

  floatx8 acc[4] = {};
  const unsigned short* ap0 = HS + (size_t)(r0 + lm) * 1024;
  for (int k0 = 0; k0 < 1024; k0 += 32) {
    FragB16 a;
    const unsigned short* ap = ap0 + k0 + lh * 8;
    a.q[0] = *(const int4*)ap;
    a.q[1] = *(const int4*)(ap + 16);
#pragma unroll
    for (int tt = 0; tt < 4; ++tt) {
      FragB16 b;
      const unsigned short* bp = W + (size_t)(nb + tt * 16 + lm) * 1024 + k0 + lh * 8;
      b.q[0] = *(const int4*)bp;
      b.q[1] = *(const int4*)(bp + 16);
      acc[tt] = __builtin_amdgcn_wmma_f32_16x16x32_bf16(
          false, a.v, false, b.v, (short)0, acc[tt], false, false);
    }
  }
#pragma unroll
  for (int tt = 0; tt < 4; ++tt) {
    int n = nb + tt * 16 + lm;
    float bb = bias[n];
#pragma unroll
    for (int v = 0; v < 8; ++v) {
      int r = r0 + lh * 8 + v;
      out[(size_t)r * 512 + n] = acc[tt][v] + bb;
    }
  }
}

// ---------------------------------------------------------------- launcher
extern "C" void kernel_launch(void* const* d_in, const int* in_sizes, int n_in,
                              void* d_out, int out_size, void* d_ws, size_t ws_size,
                              hipStream_t stream) {
  (void)in_sizes; (void)n_in; (void)out_size; (void)ws_size;
  const float* x     = (const float*)d_in[0];
  const float* W_ih  = (const float*)d_in[1];
  const float* W_hh  = (const float*)d_in[2];
  const float* b_ih  = (const float*)d_in[3];
  const float* b_hh  = (const float*)d_in[4];
  const float* W_key = (const float*)d_in[5];
  const float* b_key = (const float*)d_in[6];
  const float* W_val = (const float*)d_in[7];
  const float* b_val = (const float*)d_in[8];
  float* out = (float*)d_out;

  char* ws = (char*)d_ws;
  size_t off = 0;
  auto alloc = [&](size_t bytes) -> char* {
    char* p = ws + off;
    off += (bytes + 255) & ~(size_t)255;
    return p;
  };
  unsigned short* xbf  = (unsigned short*)alloc((size_t)32 * 1024 * 512 * 2);
  unsigned short* wihb = (unsigned short*)alloc((size_t)4096 * 512 * 2);
  unsigned short* whhb = (unsigned short*)alloc((size_t)4096 * 1024 * 2);
  unsigned short* wkb  = (unsigned short*)alloc((size_t)512 * 1024 * 2);
  unsigned short* wvb  = (unsigned short*)alloc((size_t)512 * 1024 * 2);
  unsigned short* xg   = (unsigned short*)alloc((size_t)1024 * 4096 * 32 * 2);
  unsigned short* hs   = (unsigned short*)alloc((size_t)32 * 1024 * 1024 * 2);
  unsigned*       bar  = (unsigned*)alloc(256);

  k_f32_to_bf16<<<4096, 256, 0, stream>>>(x, xbf, 32 * 1024 * 512);
  k_f32_to_bf16<<<2048, 256, 0, stream>>>(W_ih, wihb, 4096 * 512);
  k_f32_to_bf16<<<2048, 256, 0, stream>>>(W_hh, whhb, 4096 * 1024);
  k_f32_to_bf16<<<1024, 256, 0, stream>>>(W_key, wkb, 512 * 1024);
  k_f32_to_bf16<<<1024, 256, 0, stream>>>(W_val, wvb, 512 * 1024);
  k_init<<<1, 64, 0, stream>>>(bar);

  k_gemm_xg<<<16384, 256, 0, stream>>>(xbf, wihb, b_ih, b_hh, xg);
  k_lstm_scan<<<32, 128, 0, stream>>>(xg, whhb, hs, bar);
  k_gemm_out<<<2048, 256, 0, stream>>>(hs, wkb, b_key, out);
  k_gemm_out<<<2048, 256, 0, stream>>>(hs, wvb, b_val, out + (size_t)32 * 1024 * 512);
}
